// MultiHeadAttention_55954833932661
// MI455X (gfx1250) — compile-verified, matrix operand order checked by reading, not run on hardware
//
#include <hip/hip_runtime.h>

// ---------------------------------------------------------------------------
// MI455X (gfx1250) multi-head attention block, bf16 WMMA everywhere.
// Roofline: ~69 GFLOP vs ~1.6 GB HBM traffic (536MB atten tensor written,
// then re-read by softmax and the context GEMM) -> HBM-bound (~70us @ 23.3
// TB/s). bf16 WMMA (16x16x32) keeps compute far below that floor; fp32 WMMA
// (16x16x4) would not. Register tiling: each wave owns a 16x64 output strip
// (4 f32 accumulators) and reuses one A fragment across 4 WMMAs per k-step,
// so the XDL pipe stays fed while bf16 converts co-execute on the VALU.
// All intermediates live in L2 (192 MB) -- total working set ~56 MB.
// ---------------------------------------------------------------------------

typedef __attribute__((ext_vector_type(16))) __bf16 v16bf;
typedef __attribute__((ext_vector_type(8)))  __bf16 v8bf;
typedef __attribute__((ext_vector_type(8)))  float  v8f;
typedef __attribute__((ext_vector_type(4)))  float  v4f;

#define D_MODEL 1024
#define N_HEADS 16
#define D_HID   64
#define BATCH   2
#define SEQ     2048
#define NEGINF  (-1e9f)
#define LN_EPS  1e-5f

// ---- fragment loaders -----------------------------------------------------
// 16-bit A/B operand (16x32 / 32x16): lane&15 selects row/col, element i maps
// K = base + i (i<8) and K = base+16+(i-8) (i>=8), base = (lane&16)?8:0.

__device__ __forceinline__ v16bf frag_from_f32(const float* __restrict__ row,
                                               int kk, int lane) {
  const int off = (lane & 16) ? 8 : 0;
  const float* p = row + kk + off;
  v4f x0 = *(const v4f*)(p);
  v4f x1 = *(const v4f*)(p + 4);
  v4f x2 = *(const v4f*)(p + 16);
  v4f x3 = *(const v4f*)(p + 20);
  v16bf a;
#pragma unroll
  for (int i = 0; i < 4; ++i) {
    a[i]      = (__bf16)x0[i];
    a[4 + i]  = (__bf16)x1[i];
    a[8 + i]  = (__bf16)x2[i];
    a[12 + i] = (__bf16)x3[i];
  }
  return a;
}

__device__ __forceinline__ v16bf frag_from_bf16(const __bf16* __restrict__ row,
                                                int kk, int lane) {
  const int off = (lane & 16) ? 8 : 0;
  v8bf b0 = *(const v8bf*)(row + kk + off);        // 16B aligned
  v8bf b1 = *(const v8bf*)(row + kk + off + 16);
  v16bf b;
#pragma unroll
  for (int i = 0; i < 8; ++i) { b[i] = b0[i]; b[8 + i] = b1[i]; }
  return b;
}

#define WMMA_BF16(a, b, c) \
  __builtin_amdgcn_wmma_f32_16x16x32_bf16(false, (a), false, (b), (short)0, (c), false, false)

// ---- 1) pack weights fp32 [K][N] -> bf16 [N][K] (K-major, contiguous B-frag)
__global__ __launch_bounds__(256) void pack_weights_kernel(
    const float* __restrict__ Wq, const float* __restrict__ Wk,
    const float* __restrict__ Wv, const float* __restrict__ Wo,
    __bf16* __restrict__ Wt) {  // [4][1024][1024]
  size_t idx = (size_t)blockIdx.x * 256 + threadIdx.x;
  int w   = (int)(idx >> 20);
  int rem = (int)(idx & 0xFFFFF);
  int n = rem >> 10;
  int k = rem & 1023;
  const float* W = (w == 0) ? Wq : (w == 1) ? Wk : (w == 2) ? Wv : Wo;
  Wt[idx] = (__bf16)W[(size_t)k * D_MODEL + n];
}

// ---- 2) fused QKV projection: X[4096,1024] @ Wt^T -> bf16, head-split ------
// Each wave: 16(M) x 64(N) strip = 4 accumulators, A fragment reused 4x.
// z==0: q -> [B][H][L][64]; z==1: k -> [B][H][L][64]; z==2: v -> [B][H][64][L]
__global__ __launch_bounds__(256) void qkv_proj_kernel(
    const float* __restrict__ Xq, const float* __restrict__ Xk,
    const float* __restrict__ Xv,
    const __bf16* __restrict__ Wtq, const __bf16* __restrict__ Wtk,
    const __bf16* __restrict__ Wtv,
    const float* __restrict__ bq, const float* __restrict__ bk,
    const float* __restrict__ bv,
    __bf16* __restrict__ Dq, __bf16* __restrict__ Dk, __bf16* __restrict__ Dv) {
  const int z = blockIdx.z;
  const float*  X    = (z == 0) ? Xq  : (z == 1) ? Xk  : Xv;
  const __bf16* Wt   = (z == 0) ? Wtq : (z == 1) ? Wtk : Wtv;
  const float*  bias = (z == 0) ? bq  : (z == 1) ? bk  : bv;
  __bf16*       D    = (z == 0) ? Dq  : (z == 1) ? Dk  : Dv;

  const int lane = threadIdx.x & 31;
  const int wave = threadIdx.x >> 5;
  const int m0 = blockIdx.x * 16;
  const int n0 = blockIdx.y * 512 + wave * 64;

  const float*  aptr = X  + (size_t)(m0 + (lane & 15)) * D_MODEL;
  const __bf16* bptr = Wt + (size_t)(n0 + (lane & 15)) * D_MODEL;

  v8f acc[4] = {};
  for (int kk = 0; kk < D_MODEL; kk += 32) {
    __builtin_prefetch(aptr + kk + 64, 0, 1);     // global_prefetch_b8
    v16bf a = frag_from_f32(aptr, kk, lane);
#pragma unroll
    for (int j = 0; j < 4; ++j) {
      v16bf b = frag_from_bf16(bptr + (size_t)j * 16 * D_MODEL, kk, lane);
      acc[j] = WMMA_BF16(a, b, acc[j]);
    }
  }

#pragma unroll
  for (int j = 0; j < 4; ++j) {
    const int ncol = n0 + j * 16 + (lane & 15);
    const float bn = bias[ncol];
    const int h = ncol >> 6, d = ncol & 63;
#pragma unroll
    for (int r = 0; r < 8; ++r) {
      int gm = m0 + r + ((lane & 16) ? 8 : 0);
      int bi = gm / SEQ, l = gm % SEQ;
      float v = acc[j][r] + bn;
      if (z < 2)
        D[(((size_t)bi * N_HEADS + h) * SEQ + l) * D_HID + d] = (__bf16)v;
      else
        D[(((size_t)bi * N_HEADS + h) * D_HID + d) * SEQ + l] = (__bf16)v;
    }
  }
}

// ---- 3) scores = (Qh @ Kh^T) * scale, masked -> atten (fp32, pre-softmax) --
// Each wave: 16 queries x 64 keys (4 accumulators), K-loop = 64 (2 steps).
__global__ __launch_bounds__(256) void scores_kernel(
    const __bf16* __restrict__ Qh, const __bf16* __restrict__ Kh,
    const unsigned char* __restrict__ mask, float* __restrict__ atten) {
  const int lane = threadIdx.x & 31;
  const int wave = threadIdx.x >> 5;
  const int z  = blockIdx.z;              // b*H + h
  const int bi = z / N_HEADS;
  const int m0 = blockIdx.x * 16;                 // query tile
  const int n0 = blockIdx.y * 512 + wave * 64;    // key strip

  const __bf16* aptr = Qh + ((size_t)z * SEQ + (m0 + (lane & 15))) * D_HID;
  const __bf16* bptr = Kh + ((size_t)z * SEQ + (n0 + (lane & 15))) * D_HID;

  v8f acc[4] = {};
#pragma unroll
  for (int kk = 0; kk < D_HID; kk += 32) {
    v16bf a = frag_from_bf16(aptr, kk, lane);
#pragma unroll
    for (int j = 0; j < 4; ++j) {
      v16bf b = frag_from_bf16(bptr + (size_t)j * 16 * D_HID, kk, lane);
      acc[j] = WMMA_BF16(a, b, acc[j]);
    }
  }

  const float scale = 0.125f;   // 1/sqrt(64)
#pragma unroll
  for (int j = 0; j < 4; ++j) {
    const int kcol = n0 + j * 16 + (lane & 15);
#pragma unroll
    for (int r = 0; r < 8; ++r) {
      int q = m0 + r + ((lane & 16) ? 8 : 0);
      float s = acc[j][r] * scale;
      unsigned char mv = mask[((size_t)bi * SEQ + q) * SEQ + kcol];
      atten[((size_t)z * SEQ + q) * SEQ + kcol] = mv ? NEGINF : s;
    }
  }
}

// ---- 4) row softmax over L=2048, in place ----------------------------------
__global__ __launch_bounds__(256) void softmax_kernel(float* __restrict__ atten) {
  __shared__ float red[256];
  float* p = atten + (size_t)blockIdx.x * SEQ;
  const int t = threadIdx.x;
  float v[8];
  float mx = -3.4e38f;
#pragma unroll
  for (int i = 0; i < 8; ++i) { v[i] = p[t + i * 256]; mx = fmaxf(mx, v[i]); }
  red[t] = mx; __syncthreads();
  for (int s = 128; s > 0; s >>= 1) {
    if (t < s) red[t] = fmaxf(red[t], red[t + s]);
    __syncthreads();
  }
  mx = red[0]; __syncthreads();
  float sum = 0.f;
#pragma unroll
  for (int i = 0; i < 8; ++i) { v[i] = __expf(v[i] - mx); sum += v[i]; }
  red[t] = sum; __syncthreads();
  for (int s = 128; s > 0; s >>= 1) {
    if (t < s) red[t] += red[t + s];
    __syncthreads();
  }
  float inv = 1.0f / red[0];
#pragma unroll
  for (int i = 0; i < 8; ++i) p[t + i * 256] = v[i] * inv;
}

// ---- 5) context = P @ V : one wave covers the full 64-wide head -------------
__global__ __launch_bounds__(256) void context_kernel(
    const float* __restrict__ atten, const __bf16* __restrict__ VT,
    __bf16* __restrict__ ctx) {
  const int lane = threadIdx.x & 31;
  const int wave = threadIdx.x >> 5;
  const int z  = blockIdx.y;            // b*H + h
  const int bi = z / N_HEADS, h = z % N_HEADS;
  const int m0 = blockIdx.x * 128 + wave * 16;

  const float*  aptr = atten + ((size_t)z * SEQ + (m0 + (lane & 15))) * SEQ;
  const __bf16* bptr = VT + ((size_t)z * D_HID + (lane & 15)) * SEQ;

  v8f acc[4] = {};
  for (int kk = 0; kk < SEQ; kk += 32) {
    __builtin_prefetch(aptr + kk + 64, 0, 1);
    v16bf a = frag_from_f32(aptr, kk, lane);
#pragma unroll
    for (int j = 0; j < 4; ++j) {
      v16bf b = frag_from_bf16(bptr + (size_t)j * 16 * SEQ, kk, lane);
      acc[j] = WMMA_BF16(a, b, acc[j]);
    }
  }

#pragma unroll
  for (int j = 0; j < 4; ++j) {
    const int d = j * 16 + (lane & 15);
#pragma unroll
    for (int r = 0; r < 8; ++r) {
      int q = m0 + r + ((lane & 16) ? 8 : 0);
      size_t gm = (size_t)bi * SEQ + q;
      ctx[gm * (N_HEADS * D_HID) + h * D_HID + d] = (__bf16)acc[j][r];
    }
  }
}

// ---- 6) out = ctx @ Wo + bo + Q (residual) -> xbuf fp32 ---------------------
__global__ __launch_bounds__(256) void outproj_kernel(
    const __bf16* __restrict__ ctx, const __bf16* __restrict__ WtO,
    const float* __restrict__ bo, const float* __restrict__ Qin,
    float* __restrict__ xbuf) {
  const int lane = threadIdx.x & 31;
  const int wave = threadIdx.x >> 5;
  const int m0 = blockIdx.x * 16;
  const int n0 = blockIdx.y * 512 + wave * 64;

  const __bf16* aptr = ctx + (size_t)(m0 + (lane & 15)) * D_MODEL;
  const __bf16* bptr = WtO + (size_t)(n0 + (lane & 15)) * D_MODEL;

  v8f acc[4] = {};
  for (int kk = 0; kk < D_MODEL; kk += 32) {
    v16bf a = frag_from_bf16(aptr, kk, lane);
#pragma unroll
    for (int j = 0; j < 4; ++j) {
      v16bf b = frag_from_bf16(bptr + (size_t)j * 16 * D_MODEL, kk, lane);
      acc[j] = WMMA_BF16(a, b, acc[j]);
    }
  }

#pragma unroll
  for (int j = 0; j < 4; ++j) {
    const int n = n0 + j * 16 + (lane & 15);
    const float bn = bo[n];
#pragma unroll
    for (int r = 0; r < 8; ++r) {
      int gm = m0 + r + ((lane & 16) ? 8 : 0);
      size_t o = (size_t)gm * D_MODEL + n;
      xbuf[o] = acc[j][r] + bn + Qin[o];
    }
  }
}

// ---- 7) LayerNorm rows of xbuf -> y ----------------------------------------
__global__ __launch_bounds__(256) void layernorm_kernel(
    const float* __restrict__ xbuf, const float* __restrict__ gamma,
    const float* __restrict__ beta, float* __restrict__ y) {
  __shared__ float red[256];
  const float* x = xbuf + (size_t)blockIdx.x * D_MODEL;
  const int t = threadIdx.x;
  float v[4];
  float s = 0.f;
#pragma unroll
  for (int i = 0; i < 4; ++i) { v[i] = x[t + i * 256]; s += v[i]; }
  red[t] = s; __syncthreads();
  for (int st = 128; st > 0; st >>= 1) {
    if (t < st) red[t] += red[t + st];
    __syncthreads();
  }
  const float mu = red[0] * (1.0f / D_MODEL);
  __syncthreads();
  float s2 = 0.f;
#pragma unroll
  for (int i = 0; i < 4; ++i) { float d = v[i] - mu; s2 += d * d; }
  red[t] = s2; __syncthreads();
  for (int st = 128; st > 0; st >>= 1) {
    if (t < st) red[t] += red[t + st];
    __syncthreads();
  }
  const float inv = rsqrtf(red[0] * (1.0f / D_MODEL) + LN_EPS);
  float* out = y + (size_t)blockIdx.x * D_MODEL;
#pragma unroll
  for (int i = 0; i < 4; ++i) {
    int n = t + i * 256;
    out[n] = (v[i] - mu) * inv * gamma[n] + beta[n];
  }
}

// ---------------------------------------------------------------------------
extern "C" void kernel_launch(void* const* d_in, const int* in_sizes, int n_in,
                              void* d_out, int out_size, void* d_ws, size_t ws_size,
                              hipStream_t stream) {
  const float* Q  = (const float*)d_in[0];
  const float* K  = (const float*)d_in[1];
  const float* V  = (const float*)d_in[2];
  const unsigned char* mask = (const unsigned char*)d_in[3];  // jnp bool = 1B
  const float* Wq = (const float*)d_in[4];
  const float* bq = (const float*)d_in[5];
  const float* Wk = (const float*)d_in[6];
  const float* bk = (const float*)d_in[7];
  const float* Wv = (const float*)d_in[8];
  const float* bv = (const float*)d_in[9];
  const float* Wo = (const float*)d_in[10];
  const float* bo = (const float*)d_in[11];
  const float* gamma = (const float*)d_in[12];
  const float* beta  = (const float*)d_in[13];

  // workspace carve-up (~56 MB)
  char* ws = (char*)d_ws;
  __bf16* wt  = (__bf16*)(ws);                       // [4][1024][1024]  8 MB
  __bf16* qb  = (__bf16*)(ws + ((size_t)8  << 20));  // [B][H][L][64]    8 MB
  __bf16* kb  = (__bf16*)(ws + ((size_t)16 << 20));  // [B][H][L][64]    8 MB
  __bf16* vT  = (__bf16*)(ws + ((size_t)24 << 20));  // [B][H][64][L]    8 MB
  __bf16* ctx = (__bf16*)(ws + ((size_t)32 << 20));  // [B*L][1024]      8 MB
  float*  xb  = (float*) (ws + ((size_t)40 << 20));  // [B*L][1024]     16 MB

  float* y     = (float*)d_out;                              // [B][L][1024]
  float* atten = y + (size_t)BATCH * SEQ * D_MODEL;          // [B][H][L][L]

  pack_weights_kernel<<<16384, 256, 0, stream>>>(Wq, Wk, Wv, Wo, wt);

  qkv_proj_kernel<<<dim3(256, 2, 3), 256, 0, stream>>>(
      Q, K, V,
      wt, wt + ((size_t)1 << 20), wt + ((size_t)2 << 20),
      bq, bk, bv,
      qb, kb, vT);

  scores_kernel<<<dim3(128, 4, BATCH * N_HEADS), 256, 0, stream>>>(
      qb, kb, mask, atten);

  softmax_kernel<<<BATCH * N_HEADS * SEQ, 256, 0, stream>>>(atten);

  context_kernel<<<dim3(16, BATCH * N_HEADS), 256, 0, stream>>>(atten, vT, ctx);

  outproj_kernel<<<dim3(256, 2), 256, 0, stream>>>(
      ctx, wt + ((size_t)3 << 20), bo, Q, xb);

  layernorm_kernel<<<BATCH * SEQ, 256, 0, stream>>>(xb, gamma, beta, y);
}
